// AttentionLSTM_80161269613041
// MI455X (gfx1250) — compile-verified, matrix operand order checked by reading, not run on hardware
//
#include <hip/hip_runtime.h>
#include <math.h>

// ---------------- problem constants ----------------
#define NB    256      // batch N
#define TT    64       // timesteps
#define DIM   1024     // input dim D
#define HD    1024     // hidden dim H
#define K3    3072     // D + 2H
#define FOURH 4096     // 4*H
#define KT_X  32       // k-tiles (of 32) in x region
#define KT_HA 64       // k-tiles in h|attn region
#define KT_ALL 96      // total k-tiles
#define FRAG  512      // bf16 elements per 16x32 (A) / 32x16 (B) fragment

typedef __bf16 bf16;
typedef __attribute__((ext_vector_type(16))) __bf16 v16bf;
typedef __attribute__((ext_vector_type(8)))  float  v8f;

// Position of element (row-in-tile, k-in-tile) inside a 512-element WMMA
// fragment laid out [lane][16] per the CDNA5 16-bit A/B VGPR layout:
//   lanes 0-15: positions 0-7 hold K=0-7,  8-15 hold K=16-23
//   lanes 16-31: positions 0-7 hold K=8-15, 8-15 hold K=24-31
__device__ __forceinline__ int frag_idx(int row16, int kk) {
  int lane = row16 + ((kk & 8) ? 16 : 0);
  int pos  = (kk & 7) + ((kk & 16) ? 8 : 0);
  return lane * 16 + pos;
}

// ---------------- one-time: x (N,T,D) fp32 -> A-fragment-swizzled bf16 ----------------
// xSw layout: [t][mtile(16)][ktile(32)][lane(32)][16]
__global__ __launch_bounds__(256) void conv_x_sw(const float* __restrict__ x,
                                                 bf16* __restrict__ xSw) {
  const long long total  = (long long)NB * TT * DIM;
  const long long stride = (long long)gridDim.x * blockDim.x;
  for (long long i = (long long)blockIdx.x * blockDim.x + threadIdx.x; i < total; i += stride) {
    int n   = (int)(i / (TT * DIM));
    int rem = (int)(i % (TT * DIM));
    int t = rem / DIM, d = rem % DIM;
    size_t dst = (((size_t)t * 16 + (n >> 4)) * KT_X + (d >> 5)) * FRAG + frag_idx(n & 15, d & 31);
    xSw[dst] = (bf16)x[i];
  }
}

// ---------------- one-time: weight (1024,4096) fp32 -> B-fragment-swizzled bf16 -------
// WSw layout: [ntile(256)][ktile(96)][lane(32)][16]; kbase selects Wx/Wh/Wattn k-range
__global__ __launch_bounds__(256) void conv_w_sw(const float* __restrict__ src,
                                                 bf16* __restrict__ WSw, int kbase) {
  const long long total  = (long long)DIM * FOURH;
  const long long stride = (long long)gridDim.x * blockDim.x;
  for (long long i = (long long)blockIdx.x * blockDim.x + threadIdx.x; i < total; i += stride) {
    int kr = (int)(i >> 12);          // row within this weight matrix
    int j  = (int)(i & (FOURH - 1));  // output column
    int k  = kbase + kr;              // row within fused K3
    size_t dst = ((size_t)(j >> 4) * KT_ALL + (k >> 5)) * FRAG + frag_idx(j & 15, k & 31);
    WSw[dst] = (bf16)src[i];
  }
}

// ---------------- h0 = c0 = mean over the 4x4 grid of A ----------------
__global__ __launch_bounds__(256) void init_state(const float* __restrict__ A,
                                                  float* __restrict__ h,
                                                  float* __restrict__ c) {
  int idx = blockIdx.x * 256 + threadIdx.x;      // 0 .. NB*HD-1
  const float* Ar = A + (size_t)idx * 16;
  float s = 0.f;
#pragma unroll
  for (int p = 0; p < 16; ++p) s += Ar[p];
  s *= (1.0f / 16.0f);
  h[idx] = s;
  c[idx] = s;
}

// ---------------- per-step: 16-slot attention; write h|attn bf16 in A-fragment order --
// actSw layout: [mtile(16)][ktile(64)][lane(32)][16]   (k2 = 0..1023 h, 1024..2047 attn)
__global__ __launch_bounds__(256) void attn_pack(const float* __restrict__ A,   // [NB][HD][16]
                                                 const float* __restrict__ h,   // [NB][HD]
                                                 bf16* __restrict__ actSw) {
  __shared__ float red[256][16];
  __shared__ float wsm[16];
  const int n   = blockIdx.x;
  const int tid = threadIdx.x;

  float s[16];
#pragma unroll
  for (int p = 0; p < 16; ++p) s[p] = 0.f;
  for (int hh = tid; hh < HD; hh += 256) {
    float hv = h[(size_t)n * HD + hh];
    const float* Ar = A + ((size_t)n * HD + hh) * 16;
#pragma unroll
    for (int p = 0; p < 16; ++p) s[p] += hv * Ar[p];
  }
#pragma unroll
  for (int p = 0; p < 16; ++p) red[tid][p] = s[p];
  __syncthreads();
  for (int st = 128; st > 0; st >>= 1) {
    if (tid < st) {
#pragma unroll
      for (int p = 0; p < 16; ++p) red[tid][p] += red[tid + st][p];
    }
    __syncthreads();
  }
  if (tid == 0) {
    float e[16];
    float mx = -INFINITY;
#pragma unroll
    for (int p = 0; p < 16; ++p) {
      float v = red[0][p] * 0.03125f;   // 1/sqrt(1024)
      e[p] = v;
      mx = fmaxf(mx, v);
    }
    float sum = 0.f;
#pragma unroll
    for (int p = 0; p < 16; ++p) { e[p] = __expf(e[p] - mx); sum += e[p]; }
    float inv = 1.0f / sum;
#pragma unroll
    for (int p = 0; p < 16; ++p) wsm[p] = e[p] * inv;
  }
  __syncthreads();

  const int mt = n >> 4, mr = n & 15;
  bf16* base = actSw + (size_t)mt * KT_HA * FRAG;
  for (int hh = tid; hh < HD; hh += 256) {
    const float* Ar = A + ((size_t)n * HD + hh) * 16;
    float a = 0.f;
#pragma unroll
    for (int p = 0; p < 16; ++p) a += Ar[p] * wsm[p];
    int k2h = hh;             // h region
    int k2a = HD + hh;        // attn region
    base[(size_t)(k2h >> 5) * FRAG + frag_idx(mr, k2h & 31)] = (bf16)h[(size_t)n * HD + hh];
    base[(size_t)(k2a >> 5) * FRAG + frag_idx(mr, k2a & 31)] = (bf16)a;
  }
}

// ---------------- per-step fused GEMM: gates = [xt|h|attn](256xK3) @ Wcat(K3x4096) ----
// LDS-free: all fragments are pre-swizzled in memory (L2-resident weights).
// 128 threads = 4 waves; each wave owns a 32(M)x64(N) strip = 2x4 accumulators,
// 8 v_wmma_f32_16x16x32_bf16 per K-step fed by 6 coalesced 32B fragment loads.
__global__ __launch_bounds__(128) void gemm_gates(const bf16* __restrict__ xSw,
                                                  const bf16* __restrict__ actSw,
                                                  const bf16* __restrict__ WSw,
                                                  float* __restrict__ gates,
                                                  int t) {
  const int wave = threadIdx.x >> 5;
  const int lane = threadIdx.x & 31;
  const int mt0  = blockIdx.y * 2;                 // two M-tiles per wave
  const int nt0  = blockIdx.x * 16 + wave * 4;     // four N-tiles per wave
  const int lo   = lane * 16;                      // per-lane fragment offset (bf16)

  v8f acc[2][4] = {};

  // K phase 1: x region (ktile 0..31), A from xSw[t]
  {
    const bf16* a0p = xSw + ((size_t)t * 16 + mt0)     * KT_X * FRAG;
    const bf16* a1p = xSw + ((size_t)t * 16 + mt0 + 1) * KT_X * FRAG;
    for (int kt = 0; kt < KT_X; ++kt) {
      v16bf a0 = *(const v16bf*)(a0p + (size_t)kt * FRAG + lo);
      v16bf a1 = *(const v16bf*)(a1p + (size_t)kt * FRAG + lo);
#pragma unroll
      for (int j = 0; j < 4; ++j) {
        v16bf b = *(const v16bf*)(WSw + ((size_t)(nt0 + j) * KT_ALL + kt) * FRAG + lo);
        acc[0][j] = __builtin_amdgcn_wmma_f32_16x16x32_bf16(false, a0, false, b, (short)0, acc[0][j], false, false);
        acc[1][j] = __builtin_amdgcn_wmma_f32_16x16x32_bf16(false, a1, false, b, (short)0, acc[1][j], false, false);
      }
    }
  }
  // K phase 2: h|attn region (ktile 32..95), A from actSw
  {
    const bf16* a0p = actSw + (size_t)mt0       * KT_HA * FRAG;
    const bf16* a1p = actSw + (size_t)(mt0 + 1) * KT_HA * FRAG;
    for (int kt = 0; kt < KT_HA; ++kt) {
      v16bf a0 = *(const v16bf*)(a0p + (size_t)kt * FRAG + lo);
      v16bf a1 = *(const v16bf*)(a1p + (size_t)kt * FRAG + lo);
#pragma unroll
      for (int j = 0; j < 4; ++j) {
        v16bf b = *(const v16bf*)(WSw + ((size_t)(nt0 + j) * KT_ALL + KT_X + kt) * FRAG + lo);
        acc[0][j] = __builtin_amdgcn_wmma_f32_16x16x32_bf16(false, a0, false, b, (short)0, acc[0][j], false, false);
        acc[1][j] = __builtin_amdgcn_wmma_f32_16x16x32_bf16(false, a1, false, b, (short)0, acc[1][j], false, false);
      }
    }
  }

  // Epilogue: C layout — VGPR r holds M=r (lanes 0-15) / M=r+8 (lanes 16-31), N=lane&15
#pragma unroll
  for (int i = 0; i < 2; ++i) {
#pragma unroll
    for (int j = 0; j < 4; ++j) {
#pragma unroll
      for (int r = 0; r < 8; ++r) {
        int m = (mt0 + i) * 16 + ((lane < 16) ? r : (r + 8));
        gates[(size_t)m * FOURH + (nt0 + j) * 16 + (lane & 15)] = acc[i][j][r];
      }
    }
  }
}

// ---------------- per-step LSTM pointwise update ----------------
__global__ __launch_bounds__(256) void lstm_step(const float* __restrict__ gates,
                                                 const float* __restrict__ b,
                                                 float* __restrict__ hbuf,
                                                 float* __restrict__ cbuf,
                                                 float* __restrict__ out,  // [NB][TT][HD]
                                                 int t) {
  int idx = blockIdx.x * 256 + threadIdx.x;      // 0 .. NB*HD-1
  int n = idx >> 10;
  int j = idx & (HD - 1);
  const float* g = gates + (size_t)n * FOURH;
  float iv = g[j]          + b[j];
  float fv = g[j + HD]     + b[j + HD];
  float ov = g[j + 2 * HD] + b[j + 2 * HD];
  float gv = g[j + 3 * HD] + b[j + 3 * HD];
  float si = 1.f / (1.f + __expf(-iv));
  float sf = 1.f / (1.f + __expf(-fv));
  float so = 1.f / (1.f + __expf(-ov));
  float cn = sf * cbuf[idx] + si * tanhf(gv);
  float hn = so * tanhf(cn);
  cbuf[idx] = cn;
  hbuf[idx] = hn;
  out[((size_t)n * TT + t) * HD + j] = hn;
}

// ---------------- host launch ----------------
extern "C" void kernel_launch(void* const* d_in, const int* in_sizes, int n_in,
                              void* d_out, int out_size, void* d_ws, size_t ws_size,
                              hipStream_t stream) {
  const float* x     = (const float*)d_in[0];   // (N,T,D)
  const float* A     = (const float*)d_in[1];   // (N,H,4,4) == (N,H,16)
  const float* Wx    = (const float*)d_in[2];   // (D,4H)
  const float* Wh    = (const float*)d_in[3];   // (H,4H)
  const float* Wattn = (const float*)d_in[4];   // (H,4H)
  const float* bias  = (const float*)d_in[5];   // (4H)
  float* out = (float*)d_out;                   // (N,T,H)

  // workspace carve-out (~66 MB)
  char* ws = (char*)d_ws;
  bf16* WSw   = (bf16*)ws;  ws += (size_t)K3 * FOURH * sizeof(bf16);        // 25.2 MB
  bf16* xSw   = (bf16*)ws;  ws += (size_t)NB * TT * DIM * sizeof(bf16);     // 33.6 MB
  bf16* actSw = (bf16*)ws;  ws += (size_t)NB * 2 * HD * sizeof(bf16);       // 1.0 MB
  float* gates = (float*)ws; ws += (size_t)NB * FOURH * sizeof(float);      // 4.2 MB
  float* hbuf  = (float*)ws; ws += (size_t)NB * HD * sizeof(float);         // 1 MB
  float* cbuf  = (float*)ws; ws += (size_t)NB * HD * sizeof(float);         // 1 MB

  // one-time: swizzle weights into B-fragment order, x into A-fragment order
  conv_w_sw<<<1024, 256, 0, stream>>>(Wx,    WSw, 0);
  conv_w_sw<<<1024, 256, 0, stream>>>(Wh,    WSw, DIM);
  conv_w_sw<<<1024, 256, 0, stream>>>(Wattn, WSw, 2 * DIM);
  conv_x_sw<<<2048, 256, 0, stream>>>(x, xSw);
  init_state<<<(NB * HD) / 256, 256, 0, stream>>>(A, hbuf, cbuf);

  // recurrent scan: 3 kernels per step
  dim3 ggrid(FOURH / 256, NB / 32);   // 16 x 8 blocks, 128 thr (4 waves)
  for (int t = 0; t < TT; ++t) {
    attn_pack<<<NB, 256, 0, stream>>>(A, hbuf, actSw);
    gemm_gates<<<ggrid, 128, 0, stream>>>(xSw, actSw, WSw, gates, t);
    lstm_step<<<(NB * HD) / 256, 256, 0, stream>>>(gates, bias, hbuf, cbuf, out, t);
  }
}
